// LlavaViTDecoder_12000138625585
// MI455X (gfx1250) — matrix operand order repaired and checked by reading, compiled, not run on hardware
//
#include <hip/hip_runtime.h>
#include <hip/hip_bf16.h>

typedef _Float16 v16h __attribute__((ext_vector_type(16)));
typedef float    v8f  __attribute__((ext_vector_type(8)));
typedef _Float16 h2   __attribute__((ext_vector_type(2)));
typedef _Float16 h8   __attribute__((ext_vector_type(8)));

#define B_    2
#define HP_   16
#define WP_   16
#define L_    1024
#define NVIS_ 512
#define D_    384
#define HD_   48
#define NH_   8
#define FF_   1536
#define NL_   8
#define HDP_  64      // head dim padded to 64 for K=32 WMMA chunks
#define FRAME_ 256    // tokens per temporal frame (HP*WP)

// ---------------------------------------------------------------------------
// WMMA fragment helpers (wave32, v_wmma_f32_16x16x32_f16)
// A layout (16x32, MxK): lane = {m = lane&15, half = lane>>4};
//   VGPR v holds K = (v<4?0:16) + half*8 + (v&3)*2 , +1
// B layout (32x16, KxN): lane supplies col n = lane&15; element e -> K as in A (half*8 + e, 16 + half*8 + e-8).
// C/D layout (16x16 f32): lane col n = lane&15; VGPR r -> row m = r + 8*(lane>>4)
// ---------------------------------------------------------------------------
__device__ __forceinline__ v16h frag_a(const _Float16* __restrict__ p, int ld) {
  int lane = threadIdx.x & 31;
  const _Float16* row = p + (size_t)(lane & 15) * ld + ((lane >> 4) << 3);
  v16h f;
#pragma unroll
  for (int v = 0; v < 8; ++v) {
    int k = ((v & 4) << 2) + ((v & 3) << 1);   // {0,2,4,6,16,18,20,22}
    h2 t = *(const h2*)(row + k);
    f[2 * v]     = t[0];
    f[2 * v + 1] = t[1];
  }
  return f;
}

__device__ __forceinline__ v16h frag_b(const _Float16* __restrict__ p, int ld) {
  int lane = threadIdx.x & 31;
  const _Float16* col = p + (lane & 15) + (size_t)((lane >> 4) << 3) * ld;
  v16h f;
#pragma unroll
  for (int e = 0; e < 8; ++e) { f[e] = col[(size_t)e * ld]; f[8 + e] = col[(size_t)(16 + e) * ld]; }
  return f;
}

__device__ __forceinline__ v8f wmma16(v16h a, v16h b, v8f c) {
  return __builtin_amdgcn_wmma_f32_16x16x32_f16(false, a, false, b,
                                                (short)0, c, false, false);
}

// ---------------------------------------------------------------------------
// WMMA GEMM, 16x64 strip per wave (4 accumulators, A-fragment reused 4x),
// software-pipelined: fragments for iteration k+32 are prefetched while the
// WMMAs consume iteration k. The prefetched values are live across the back
// edge -> regalloc must double-buffer them, so the loadcnt waits move ahead
// of the *next* iteration's WMMAs and load latency hides behind matrix ops.
// grid = (M/16, N/64), block = 1 wave. All operands L2-resident.
// ---------------------------------------------------------------------------
__global__ void gemm_wmma(const _Float16* __restrict__ A, int lda,
                          const _Float16* __restrict__ Bw, int ldb,
                          const float* __restrict__ bias,
                          const float* __restrict__ Cres,
                          float* __restrict__ outF,
                          _Float16* __restrict__ outH,
                          int ldc, int K, int act_gelu)
{
  int tm = blockIdx.x, tn = blockIdx.y;
  const _Float16* Ab = A + (size_t)tm * 16 * lda;
  const _Float16* Bb = Bw + tn * 64;
  v8f c0 = {}, c1 = {}, c2 = {}, c3 = {};

  v16h fa  = frag_a(Ab, lda);
  v16h fb0 = frag_b(Bb +  0, ldb);
  v16h fb1 = frag_b(Bb + 16, ldb);
  v16h fb2 = frag_b(Bb + 32, ldb);
  v16h fb3 = frag_b(Bb + 48, ldb);

  for (int k0 = 0; k0 < K; k0 += 32) {
    int k1 = (k0 + 32 < K) ? (k0 + 32) : 0;      // clamped (branchless) prefetch
    const _Float16* bp = Bb + (size_t)k1 * ldb;
    v16h na  = frag_a(Ab + k1, lda);
    v16h nb0 = frag_b(bp +  0, ldb);
    v16h nb1 = frag_b(bp + 16, ldb);
    v16h nb2 = frag_b(bp + 32, ldb);
    v16h nb3 = frag_b(bp + 48, ldb);

    c0 = wmma16(fa, fb0, c0);
    c1 = wmma16(fa, fb1, c1);
    c2 = wmma16(fa, fb2, c2);
    c3 = wmma16(fa, fb3, c3);

    fa = na; fb0 = nb0; fb1 = nb1; fb2 = nb2; fb3 = nb3;
  }

  v8f c[4] = {c0, c1, c2, c3};
  int lane = threadIdx.x & 31;
  int n = lane & 15, half = lane >> 4;
#pragma unroll
  for (int t = 0; t < 4; ++t) {
    int col = tn * 64 + t * 16 + n;
    float bv = bias ? bias[col] : 0.0f;
#pragma unroll
    for (int r = 0; r < 8; ++r) {
      int row = tm * 16 + r + (half << 3);
      float v = c[t][r] + bv;
      if (act_gelu) v = 0.5f * v * (1.0f + erff(v * 0.70710678118f));
      if (Cres) v += Cres[(size_t)row * ldc + col];
      if (outF) outF[(size_t)row * ldc + col] = v;
      if (outH) outH[(size_t)row * ldc + col] = (_Float16)v;
    }
  }
}

// ---------------------------------------------------------------------------
// RMSNorm: one wave per token row (D=384 -> 12 elems/lane)
// ---------------------------------------------------------------------------
__global__ void rms_kernel(const float* __restrict__ x, const float* __restrict__ w,
                           _Float16* __restrict__ o16, float* __restrict__ o32)
{
  int row = blockIdx.x;
  int lane = threadIdx.x;
  const float* xr = x + (size_t)row * D_;
  float ss = 0.f;
  for (int d = lane; d < D_; d += 32) { float v = xr[d]; ss += v * v; }
  for (int off = 16; off > 0; off >>= 1) ss += __shfl_xor(ss, off, 32);
  float inv = rsqrtf(ss * (1.0f / D_) + 1e-6f);
  for (int d = lane; d < D_; d += 32) {
    float v = xr[d] * inv * w[d];
    if (o16) o16[(size_t)row * D_ + d] = (_Float16)v;
    if (o32) o32[(size_t)row * D_ + d] = v;
  }
}

// ---------------------------------------------------------------------------
// Build input sequence: unshuffle visible tokens + mask tokens
// ---------------------------------------------------------------------------
__global__ void build_x(const float* __restrict__ vis, const float* __restrict__ mtok,
                        const int* __restrict__ idr, float* __restrict__ x)
{
  int i = blockIdx.x * blockDim.x + threadIdx.x;
  if (i >= B_ * L_ * D_) return;
  int d = i % D_; int bl = i / D_; int l = bl % L_; int b = bl / L_;
  int idx = idr[b * L_ + l];
  x[i] = (idx < NVIS_) ? vis[((size_t)b * NVIS_ + idx) * D_ + d] : mtok[d];
}

// ---------------------------------------------------------------------------
// 3D RoPE tables: cos/sin [L][48]; dt=12 (6 freqs), dh=dw=18 (9 freqs each)
// ---------------------------------------------------------------------------
__global__ void rope_tables(float* __restrict__ cosT, float* __restrict__ sinT)
{
  int i = blockIdx.x * blockDim.x + threadIdx.x;
  if (i >= L_ * 24) return;
  int j = i % 24, l = i / 24;
  int ti = l / (HP_ * WP_);
  int rem = l % (HP_ * WP_);
  int hi = rem / WP_, wi = rem % WP_;
  float f;
  if (j < 6)       f = ti * __powf(10000.0f, -(2.0f * j) / 12.0f);
  else if (j < 15) f = hi * __powf(10000.0f, -(2.0f * (j - 6)) / 18.0f);
  else             f = wi * __powf(10000.0f, -(2.0f * (j - 15)) / 18.0f);
  float c = __cosf(f), s = __sinf(f);
  cosT[l * 48 + j] = c; cosT[l * 48 + 24 + j] = c;
  sinT[l * 48 + j] = s; sinT[l * 48 + 24 + j] = s;
}

// ---------------------------------------------------------------------------
// Apply RoPE, repack for attention:
//   qp [b,h][l][64]  (A operand, zero-padded 48..63)
//   kT [b,h][d=64][L] (B operand for Q*K^T, zero rows 48..63)
//   vh [b,h][l][48]  (B operand for P*V)
// ---------------------------------------------------------------------------
__global__ void rope_pack(const float* __restrict__ q, const float* __restrict__ k,
                          const float* __restrict__ v,
                          const float* __restrict__ cosT, const float* __restrict__ sinT,
                          _Float16* __restrict__ qp, _Float16* __restrict__ kT,
                          _Float16* __restrict__ vh)
{
  int i = blockIdx.x * blockDim.x + threadIdx.x;
  if (i >= B_ * NH_ * L_ * HDP_) return;
  int j = i & 63; int t = i >> 6;
  int l = t % L_; int bh = t / L_; int h = bh % NH_;
  int b = bh / NH_;
  size_t src = ((size_t)(b * L_ + l)) * D_ + h * HD_;
  _Float16 qo = (_Float16)0.f, ko = (_Float16)0.f;
  if (j < HD_) {
    float c = cosT[l * 48 + j], s = sinT[l * 48 + j];
    float qv = q[src + j], kv = k[src + j];
    float qr = (j < 24) ? -q[src + j + 24] : q[src + j - 24];
    float kr = (j < 24) ? -k[src + j + 24] : k[src + j - 24];
    qo = (_Float16)(qv * c + qr * s);
    ko = (_Float16)(kv * c + kr * s);
    vh[((size_t)bh * L_ + l) * HD_ + j] = (_Float16)v[src + j];
  }
  qp[((size_t)bh * L_ + l) * HDP_ + j] = qo;
  kT[((size_t)bh * HDP_ + j) * L_ + l] = ko;
}

// ---------------------------------------------------------------------------
// Flash attention: 1 wave per (b*h, 16-query tile). Frame-causal mask is
// exact at tile granularity (frames = 256 tokens), so masked key chunks are
// skipped entirely via kmax. Probs go through LDS to re-enter the A layout.
// ---------------------------------------------------------------------------
__global__ void attn_kernel(const _Float16* __restrict__ qp,
                            const _Float16* __restrict__ kT,
                            const _Float16* __restrict__ vh,
                            _Float16* __restrict__ o16)
{
  __shared__ _Float16 pl[16 * 32];
  int bh = blockIdx.x;
  int b = bh / NH_, h = bh % NH_;
  int tq = blockIdx.y;
  int lane = threadIdx.x & 31;
  int n = lane & 15, half = lane >> 4;

  const _Float16* qb = qp + ((size_t)bh * L_ + tq * 16) * HDP_;
  const _Float16* kb = kT + (size_t)bh * HDP_ * L_;
  const _Float16* vb = vh + (size_t)bh * L_ * HD_;

  v16h qa0 = frag_a(qb, HDP_);        // K-dims 0..31
  v16h qa1 = frag_a(qb + 32, HDP_);   // K-dims 32..63 (padded)

  v8f acc0 = {}, acc1 = {}, acc2 = {};
  float mi[8], li[8];
#pragma unroll
  for (int r = 0; r < 8; ++r) { mi[r] = -1e30f; li[r] = 0.f; }

  const float sc = 0.14433756729740643f;       // 1/sqrt(48)
  int kmax = (((tq * 16) >> 8) + 1) << 8;      // end of this query tile's frame

  for (int k0 = 0; k0 < kmax; k0 += 32) {
    // --- scores ---
    v16h kf00 = frag_b(kb + 0 * L_  + k0,      L_);
    v16h kf01 = frag_b(kb + 32 * L_ + k0,      L_);
    v16h kf10 = frag_b(kb + 0 * L_  + k0 + 16, L_);
    v16h kf11 = frag_b(kb + 32 * L_ + k0 + 16, L_);
    v8f s0 = {}, s1 = {};
    s0 = wmma16(qa0, kf00, s0);
    s0 = wmma16(qa1, kf01, s0);
    s1 = wmma16(qa0, kf10, s1);
    s1 = wmma16(qa1, kf11, s1);

    float p0[8], p1[8];
#pragma unroll
    for (int r = 0; r < 8; ++r) { p0[r] = s0[r] * sc; p1[r] = s1[r] * sc; }

#pragma unroll
    for (int r = 0; r < 8; ++r) {
      float cm = fmaxf(p0[r], p1[r]);
      for (int off = 8; off > 0; off >>= 1) cm = fmaxf(cm, __shfl_xor(cm, off, 32));
      float mn = fmaxf(mi[r], cm);
      float fac = __expf(mi[r] - mn);
      mi[r] = mn;
      p0[r] = __expf(p0[r] - mn);
      p1[r] = __expf(p1[r] - mn);
      float rs = p0[r] + p1[r];
      for (int off = 8; off > 0; off >>= 1) rs += __shfl_xor(rs, off, 32);
      li[r] = li[r] * fac + rs;
      acc0[r] *= fac; acc1[r] *= fac; acc2[r] *= fac;
      pl[(r + (half << 3)) * 32 + n]      = (_Float16)p0[r];
      pl[(r + (half << 3)) * 32 + 16 + n] = (_Float16)p1[r];
    }
    __syncthreads();
    v16h pa  = frag_a(pl, 32);
    v16h vf0 = frag_b(vb + (size_t)k0 * HD_ + 0,  HD_);
    v16h vf1 = frag_b(vb + (size_t)k0 * HD_ + 16, HD_);
    v16h vf2 = frag_b(vb + (size_t)k0 * HD_ + 32, HD_);
    acc0 = wmma16(pa, vf0, acc0);
    acc1 = wmma16(pa, vf1, acc1);
    acc2 = wmma16(pa, vf2, acc2);
    __syncthreads();
  }
#pragma unroll
  for (int r = 0; r < 8; ++r) {
    float inv = 1.0f / li[r];
    int tok = tq * 16 + r + (half << 3);
    size_t base = ((size_t)(b * L_ + tok)) * D_ + h * HD_;
    o16[base + n]      = (_Float16)(acc0[r] * inv);
    o16[base + 16 + n] = (_Float16)(acc1[r] * inv);
    o16[base + 32 + n] = (_Float16)(acc2[r] * inv);
  }
}

// ---------------------------------------------------------------------------
// fp32 -> f16 weight conversion
// ---------------------------------------------------------------------------
__global__ void cvt16(const float* __restrict__ s, _Float16* __restrict__ d, int nElem)
{
  int i = blockIdx.x * blockDim.x + threadIdx.x;
  if (i < nElem) d[i] = (_Float16)s[i];
}

// ---------------------------------------------------------------------------
// Stable unshuffle positions from ids_restore (visible first, then masked)
// ---------------------------------------------------------------------------
__global__ void pos_kernel(const int* __restrict__ idr, int* __restrict__ pos)
{
  int i = blockIdx.x * blockDim.x + threadIdx.x;
  if (i >= B_ * L_) return;
  int b = i / L_, l = i % L_;
  const int* row = idr + b * L_;
  int visv = (row[l] < NVIS_) ? 1 : 0;
  int cnt = 0;
  for (int j = 0; j < l; ++j) {
    int vj = (row[j] < NVIS_) ? 1 : 0;
    cnt += (vj == visv) ? 1 : 0;
  }
  pos[i] = visv ? cnt : NVIS_ + cnt;
}

__global__ void scatter_kernel(const float* __restrict__ xo, const int* __restrict__ pos,
                               float* __restrict__ outv, float* __restrict__ outm)
{
  int i = blockIdx.x * blockDim.x + threadIdx.x;
  if (i >= B_ * L_ * D_) return;
  int d = i % D_; int bl = i / D_; int l = bl % L_; int b = bl / L_;
  int p = pos[b * L_ + l];
  float v = xo[i];
  if (p < NVIS_) outv[((size_t)b * NVIS_ + p) * D_ + d] = v;
  else outm[((size_t)b * (L_ - NVIS_) + (p - NVIS_)) * D_ + d] = v;
}

// ---------------------------------------------------------------------------
extern "C" void kernel_launch(void* const* d_in, const int* in_sizes, int n_in,
                              void* d_out, int out_size, void* d_ws, size_t ws_size,
                              hipStream_t stream)
{
  (void)in_sizes; (void)n_in; (void)out_size; (void)ws_size;
  const float* vis  = (const float*)d_in[0];
  const float* mtok = (const float*)d_in[1];
  const float* ln1  = (const float*)d_in[2];
  const float* wq   = (const float*)d_in[3];
  const float* bq   = (const float*)d_in[4];
  const float* wk   = (const float*)d_in[5];
  const float* bk   = (const float*)d_in[6];
  const float* wv   = (const float*)d_in[7];
  const float* bv   = (const float*)d_in[8];
  const float* wo   = (const float*)d_in[9];
  const float* bo   = (const float*)d_in[10];
  const float* ln2  = (const float*)d_in[11];
  const float* w1   = (const float*)d_in[12];
  const float* b1   = (const float*)d_in[13];
  const float* w2   = (const float*)d_in[14];
  const float* b2   = (const float*)d_in[15];
  const float* lnp  = (const float*)d_in[16];
  const int*   idr  = (const int*)d_in[17];
  // d_in[18] (mask) recomputed from ids_restore; [19..21] scalars compiled in.

  char* ws = (char*)d_ws;
  size_t off = 0;
  auto alloc = [&](size_t bytes) -> char* {
    char* p = ws + off;
    off += (bytes + 255) & ~(size_t)255;
    return p;
  };

  _Float16* wq16 = (_Float16*)alloc((size_t)NL_ * D_ * D_ * 2);
  _Float16* wk16 = (_Float16*)alloc((size_t)NL_ * D_ * D_ * 2);
  _Float16* wv16 = (_Float16*)alloc((size_t)NL_ * D_ * D_ * 2);
  _Float16* wo16 = (_Float16*)alloc((size_t)NL_ * D_ * D_ * 2);
  _Float16* w116 = (_Float16*)alloc((size_t)NL_ * D_ * FF_ * 2);
  _Float16* w216 = (_Float16*)alloc((size_t)NL_ * FF_ * D_ * 2);
  float* cosT = (float*)alloc((size_t)L_ * 48 * 4);
  float* sinT = (float*)alloc((size_t)L_ * 48 * 4);
  float* xf32 = (float*)alloc((size_t)B_ * L_ * D_ * 4);
  _Float16* hD16  = (_Float16*)alloc((size_t)B_ * L_ * D_ * 2);
  _Float16* hFF16 = (_Float16*)alloc((size_t)B_ * L_ * FF_ * 2);
  float* qraw = (float*)alloc((size_t)B_ * L_ * D_ * 4);
  float* kraw = (float*)alloc((size_t)B_ * L_ * D_ * 4);
  float* vraw = (float*)alloc((size_t)B_ * L_ * D_ * 4);
  _Float16* qpB = (_Float16*)alloc((size_t)B_ * NH_ * L_ * HDP_ * 2);
  _Float16* kTB = (_Float16*)alloc((size_t)B_ * NH_ * HDP_ * L_ * 2);
  _Float16* vhB = (_Float16*)alloc((size_t)B_ * NH_ * L_ * HD_ * 2);
  _Float16* o16buf = (_Float16*)alloc((size_t)B_ * L_ * D_ * 2);
  int* posbuf = (int*)alloc((size_t)B_ * L_ * 4);

  const int TH = 256;
  auto blocks = [](int n, int t) { return (n + t - 1) / t; };

  // weights -> f16 (row-major [K][N], matches B-fragment layout)
  cvt16<<<blocks(NL_ * D_ * D_, TH), TH, 0, stream>>>(wq, wq16, NL_ * D_ * D_);
  cvt16<<<blocks(NL_ * D_ * D_, TH), TH, 0, stream>>>(wk, wk16, NL_ * D_ * D_);
  cvt16<<<blocks(NL_ * D_ * D_, TH), TH, 0, stream>>>(wv, wv16, NL_ * D_ * D_);
  cvt16<<<blocks(NL_ * D_ * D_, TH), TH, 0, stream>>>(wo, wo16, NL_ * D_ * D_);
  cvt16<<<blocks(NL_ * D_ * FF_, TH), TH, 0, stream>>>(w1, w116, NL_ * D_ * FF_);
  cvt16<<<blocks(NL_ * FF_ * D_, TH), TH, 0, stream>>>(w2, w216, NL_ * FF_ * D_);

  rope_tables<<<blocks(L_ * 24, TH), TH, 0, stream>>>(cosT, sinT);
  build_x<<<blocks(B_ * L_ * D_, TH), TH, 0, stream>>>(vis, mtok, idr, xf32);

  dim3 gD(B_ * L_ / 16, D_ / 64);    // 128 x 6
  dim3 gF(B_ * L_ / 16, FF_ / 64);   // 128 x 24

  for (int l = 0; l < NL_; ++l) {
    const _Float16* wq_l = wq16 + (size_t)l * D_ * D_;
    const _Float16* wk_l = wk16 + (size_t)l * D_ * D_;
    const _Float16* wv_l = wv16 + (size_t)l * D_ * D_;
    const _Float16* wo_l = wo16 + (size_t)l * D_ * D_;
    const _Float16* w1_l = w116 + (size_t)l * D_ * FF_;
    const _Float16* w2_l = w216 + (size_t)l * FF_ * D_;

    rms_kernel<<<B_ * L_, 32, 0, stream>>>(xf32, ln1 + l * D_, hD16, nullptr);

    gemm_wmma<<<gD, 32, 0, stream>>>(hD16, D_, wq_l, D_, bq + l * D_,
                                     nullptr, qraw, nullptr, D_, D_, 0);
    gemm_wmma<<<gD, 32, 0, stream>>>(hD16, D_, wk_l, D_, bk + l * D_,
                                     nullptr, kraw, nullptr, D_, D_, 0);
    gemm_wmma<<<gD, 32, 0, stream>>>(hD16, D_, wv_l, D_, bv + l * D_,
                                     nullptr, vraw, nullptr, D_, D_, 0);

    rope_pack<<<blocks(B_ * NH_ * L_ * HDP_, TH), TH, 0, stream>>>(
        qraw, kraw, vraw, cosT, sinT, qpB, kTB, vhB);

    attn_kernel<<<dim3(B_ * NH_, L_ / 16), 32, 0, stream>>>(qpB, kTB, vhB, o16buf);

    gemm_wmma<<<gD, 32, 0, stream>>>(o16buf, D_, wo_l, D_, bo + l * D_,
                                     xf32, xf32, nullptr, D_, D_, 0);

    rms_kernel<<<B_ * L_, 32, 0, stream>>>(xf32, ln2 + l * D_, hD16, nullptr);

    gemm_wmma<<<gF, 32, 0, stream>>>(hD16, D_, w1_l, FF_, b1 + l * FF_,
                                     nullptr, nullptr, hFF16, FF_, D_, 1);
    gemm_wmma<<<gD, 32, 0, stream>>>(hFF16, FF_, w2_l, D_, b2 + l * D_,
                                     xf32, xf32, nullptr, D_, FF_, 0);
  }

  float* outx = (float*)d_out;
  float* outv = outx + (size_t)B_ * L_ * D_;
  float* outm = outv + (size_t)B_ * NVIS_ * D_;

  rms_kernel<<<B_ * L_, 32, 0, stream>>>(xf32, lnp, nullptr, outx);
  pos_kernel<<<blocks(B_ * L_, TH), TH, 0, stream>>>(idr, posbuf);
  scatter_kernel<<<blocks(B_ * L_ * D_, TH), TH, 0, stream>>>(outx, posbuf, outv, outm);
}
